// ResetMaskLSTM_29686813950393
// MI455X (gfx1250) — matrix operand order repaired and checked by reading, not compiled, not run on hardware
//
#include <hip/hip_runtime.h>
#include <math.h>

// Problem constants (from reference): T=1024, B=32, I=512, H=512, L=2
#define T_STEPS 1024
#define BATCH   32
#define IDIM    512
#define HDIM    512
#define NL      2
#define NGATES  (4*HDIM)          // 2048
#define NWG     16                // persistent workgroups
#define NTHR    256               // 8 wave32 per WG -> 128 waves -> 128 strips of 16 cols
#define PW_THREADS (NWG*NTHR)     // 4096

typedef __attribute__((ext_vector_type(16))) __bf16 v16bf;
typedef __attribute__((ext_vector_type(8)))  __bf16 v8bf;
typedef __attribute__((ext_vector_type(8)))  float  v8f;

__device__ __forceinline__ unsigned short f2bf(float f) {
  union { float f; unsigned u; } v; v.f = f;
  unsigned r = v.u + 0x7FFFu + ((v.u >> 16) & 1u);   // round-to-nearest-even
  return (unsigned short)(r >> 16);
}
__device__ __forceinline__ float sigm(float x) { return 1.0f / (1.0f + __expf(-x)); }

// Async global->LDS copy of one 16-byte chunk (gfx1250 VGLOBAL async op,
// tracked by ASYNCcnt; no VGPR round-trip).
__device__ __forceinline__ void async_copy_b128(const unsigned short* g, unsigned short* lds_ptr) {
#if __HIP_DEVICE_COMPILE__
  unsigned lds_off = (unsigned)(unsigned long long)lds_ptr;  // LDS aperture: addr[31:0] = LDS byte offset
  asm volatile("global_load_async_to_lds_b128 %0, %1, off"
               :: "v"(lds_off), "v"((unsigned long long)g)
               : "memory");
#else
  *(v8bf*)lds_ptr = *(const v8bf*)g;
#endif
}
__device__ __forceinline__ void wait_asynccnt0() {
#if __HIP_DEVICE_COMPILE__
  asm volatile("s_wait_asynccnt 0x0" ::: "memory");
#endif
}

// ---------------- device-wide barrier (16 co-resident WGs) ----------------
__device__ __forceinline__ void grid_sync(unsigned* cnt, unsigned* gen) {
  __threadfence();
  __syncthreads();
  if (threadIdx.x == 0) {
    unsigned g = __hip_atomic_load(gen, __ATOMIC_RELAXED, __HIP_MEMORY_SCOPE_AGENT);
    unsigned a = __hip_atomic_fetch_add(cnt, 1u, __ATOMIC_ACQ_REL, __HIP_MEMORY_SCOPE_AGENT);
    if (a == NWG - 1u) {
      __hip_atomic_store(cnt, 0u, __ATOMIC_RELAXED, __HIP_MEMORY_SCOPE_AGENT);
      __hip_atomic_fetch_add(gen, 1u, __ATOMIC_ACQ_REL, __HIP_MEMORY_SCOPE_AGENT);
    } else {
      while (__hip_atomic_load(gen, __ATOMIC_ACQUIRE, __HIP_MEMORY_SCOPE_AGENT) == g) {
#if __HIP_DEVICE_COMPILE__
        __builtin_amdgcn_s_sleep(8);
#endif
      }
    }
  }
  __syncthreads();
  __threadfence();
}

// K=512 bf16 GEMM micro-loop: two 16x16 M-tiles (rows 0-15 / 16-31), one
// 16-column B strip, v_wmma_f32_16x16x32_bf16. A comes from LDS (ds_load),
// B streams from global (L2-resident weights).
// A-frag layout (16-bit A 16x32, wave32): lanes 0-15 = rows, K 0-7 & 16-23;
// lanes 16-31 = rows, K 8-15 & 24-31. B-frag: lane%16 = column, same K split
// as A (half*8 baked into bp by caller).
__device__ __forceinline__ void gemm_k512(const unsigned short* a0p,   // LDS
                                          const unsigned short* a1p,   // LDS
                                          const unsigned short* __restrict__ bp,  // global
                                          v8f& acc0, v8f& acc1) {
#pragma unroll 8
  for (int k0 = 0; k0 < 512; k0 += 32) {
    v8bf a0lo = *(const v8bf*)(a0p + k0);
    v8bf a0hi = *(const v8bf*)(a0p + k0 + 16);
    v8bf a1lo = *(const v8bf*)(a1p + k0);
    v8bf a1hi = *(const v8bf*)(a1p + k0 + 16);
    v16bf bb  = __builtin_shufflevector(*(const v8bf*)(bp + k0), *(const v8bf*)(bp + k0 + 16), 0,1,2,3,4,5,6,7,8,9,10,11,12,13,14,15);
    v16bf a0 = __builtin_shufflevector(a0lo, a0hi, 0,1,2,3,4,5,6,7,8,9,10,11,12,13,14,15);
    v16bf a1 = __builtin_shufflevector(a1lo, a1hi, 0,1,2,3,4,5,6,7,8,9,10,11,12,13,14,15);
#if __HIP_DEVICE_COMPILE__
    acc0 = __builtin_amdgcn_wmma_f32_16x16x32_bf16(false, a0, false, bb, (short)0, acc0, false, false);
    acc1 = __builtin_amdgcn_wmma_f32_16x16x32_bf16(false, a1, false, bb, (short)0, acc1, false, false);
#endif
  }
}

// ---------------- prep kernels ----------------
__global__ void cvt_bf16_kernel(const float* __restrict__ src,
                                unsigned short* __restrict__ dst, size_t n) {
  size_t i = blockIdx.x * (size_t)blockDim.x + threadIdx.x;
  size_t stride = gridDim.x * (size_t)blockDim.x;
  for (; i < n; i += stride) dst[i] = f2bf(src[i]);
}

__global__ void init_kernel(const float* __restrict__ bih, const float* __restrict__ bhh,
                            const float* __restrict__ h0,  const float* __restrict__ c0,
                            float* __restrict__ bsum, unsigned short* __restrict__ h0bf,
                            unsigned short* __restrict__ hbf, float* __restrict__ hf,
                            float* __restrict__ cst, unsigned* __restrict__ bars) {
  int i = blockIdx.x * blockDim.x + threadIdx.x;
  int stride = gridDim.x * blockDim.x;
  for (int e = i; e < NL * NGATES; e += stride) bsum[e] = bih[e] + bhh[e];
  for (int e = i; e < NL * HDIM; e += stride) h0bf[e] = f2bf(h0[e]);
  for (int e = i; e < NL * BATCH * HDIM; e += stride) {
    int l = e / (BATCH * HDIM);
    int j = e & (HDIM - 1);
    float hv = h0[l * HDIM + j];
    hbf[e] = f2bf(hv);
    hf[e]  = hv;
    cst[e] = c0[l * HDIM + j];
  }
  if (i < 2) bars[i] = 0u;
}

// ---------------- persistent LSTM kernel ----------------
__global__ __launch_bounds__(NTHR) void lstm_persistent(
    const unsigned short* __restrict__ xbf,    // [T][B][I] bf16
    const int*            __restrict__ mask,   // [T][B]
    const unsigned short* __restrict__ wihbf,  // [L][4H][I] bf16
    const unsigned short* __restrict__ whhbf,  // [L][4H][H] bf16
    const float*          __restrict__ bsum,   // [L][4H]  (b_ih + b_hh)
    const unsigned short* __restrict__ h0bf,   // [L][H] bf16
    const float*          __restrict__ c0,     // [L][H]
    unsigned short*       __restrict__ hbf,    // [L][B][H] bf16 state
    float*                __restrict__ hf,     // [L][B][H] f32 state
    float*                __restrict__ cst,    // [L][B][H] f32 state
    float*                __restrict__ gates,  // [B][4H] scratch
    unsigned* bar_cnt, unsigned* bar_gen,
    float*                __restrict__ out)    // T*B*H | h_n | c_n
{
  // LDS staging for the shared A operands (32x512 bf16 each = 32KB each).
  __shared__ unsigned short sIn[BATCH * IDIM];   // layer input (x_t or h[0])
  __shared__ unsigned short sH [BATCH * HDIM];   // reset-masked recurrent h

  const int lane = threadIdx.x & 31;
  const int wave = threadIdx.x >> 5;
  const int half = lane >> 4;        // which half-wave
  const int lcol = lane & 15;        // column within strip / row within tile
  const int strip = blockIdx.x * (NTHR / 32) + wave;   // 0..127
  const int n0 = strip * 16;                           // gate column base
  const int tid = blockIdx.x * NTHR + threadIdx.x;     // 0..4095

  const int row0 = lcol;        // M-tile 0 rows 0..15
  const int row1 = 16 + lcol;   // M-tile 1 rows 16..31

  for (int t = 0; t < T_STEPS; ++t) {
    for (int l = 0; l < NL; ++l) {
      // ======== stage A operands into LDS (async global->LDS) ========
      {
        // input A: contiguous 32KB (x_t for layer 0, h[0] for layer 1)
        const unsigned short* src = (l == 0) ? (xbf + (size_t)t * BATCH * IDIM) : hbf;
#pragma unroll
        for (int i = 0; i < 8; ++i) {
          int c = threadIdx.x + i * NTHR;          // 16B chunk id, 0..2047
          async_copy_b128(src + c * 8, sIn + c * 8);
        }
        // recurrent A: reset mask folded into per-row source selection
        const unsigned short* hl = hbf + (size_t)l * BATCH * HDIM;
        const unsigned short* hz = h0bf + l * HDIM;
#pragma unroll
        for (int i = 0; i < 8; ++i) {
          int c = threadIdx.x + i * NTHR;
          int b  = c >> 6;                          // 64 chunks per row
          int cc = c & 63;
          const unsigned short* g =
              (mask[t * BATCH + b] ? hz : (hl + (size_t)b * HDIM)) + cc * 8;
          async_copy_b128(g, sH + c * 8);
        }
        wait_asynccnt0();
      }
      __syncthreads();

      // ======== GEMM phase: gates = in @ Wih^T + h_used @ Whh^T + bias ========
      {
        float bias = bsum[l * NGATES + n0 + lcol];
        v8f acc0, acc1;
#pragma unroll
        for (int r = 0; r < 8; ++r) { acc0[r] = bias; acc1[r] = bias; }

        gemm_k512(sIn + row0 * IDIM + half * 8,
                  sIn + row1 * IDIM + half * 8,
                  wihbf + ((size_t)l * NGATES + n0 + lcol) * IDIM + half * 8,
                  acc0, acc1);
        gemm_k512(sH + row0 * HDIM + half * 8,
                  sH + row1 * HDIM + half * 8,
                  whhbf + ((size_t)l * NGATES + n0 + lcol) * HDIM + half * 8,
                  acc0, acc1);

        // C/D layout: VGPR r -> M = r + 8*half (tile0) / +16 (tile1), N = n0+lcol
        float* g = gates + n0 + lcol;
#pragma unroll
        for (int r = 0; r < 8; ++r) {
          int M = r + half * 8;
          g[(size_t)M * NGATES]        = acc0[r];
          g[(size_t)(M + 16) * NGATES] = acc1[r];
        }

        // prefetch next timestep's x slab into cache (global_prefetch_b8)
        if (l == 0 && t + 1 < T_STEPS)
          __builtin_prefetch(xbf + (size_t)(t + 1) * BATCH * IDIM + (threadIdx.x << 6), 0, 1);
      }
      grid_sync(bar_cnt, bar_gen);

      // ======== pointwise LSTM cell ========
      {
        const float* c0l = c0 + l * HDIM;
        float*          clst = cst + (size_t)l * BATCH * HDIM;
        float*          hlf  = hf  + (size_t)l * BATCH * HDIM;
        unsigned short* hlb  = hbf + (size_t)l * BATCH * HDIM;
#pragma unroll
        for (int i = 0; i < (BATCH * HDIM) / PW_THREADS; ++i) {
          int e = tid + i * PW_THREADS;      // e = b*H + j
          int b = e >> 9;
          int j = e & (HDIM - 1);
          const float* gr = gates + (size_t)b * NGATES;
          float ig = gr[j];
          float fg = gr[HDIM + j];
          float gg = gr[2 * HDIM + j];
          float og = gr[3 * HDIM + j];
          float cp = mask[t * BATCH + b] ? c0l[j] : clst[e];
          float cn = sigm(fg) * cp + sigm(ig) * tanhf(gg);
          float hn = sigm(og) * tanhf(cn);
          clst[e] = cn;
          hlf[e]  = hn;
          hlb[e]  = f2bf(hn);
          if (l == 1) out[(size_t)t * BATCH * HDIM + e] = hn;
        }
      }
      grid_sync(bar_cnt, bar_gen);
    }
  }

  // final carries: h_n then c_n appended after out[T]
  {
    float* hn = out + (size_t)T_STEPS * BATCH * HDIM;
    float* cn = hn + (size_t)NL * BATCH * HDIM;
    for (int e = tid; e < NL * BATCH * HDIM; e += PW_THREADS) {
      hn[e] = hf[e];
      cn[e] = cst[e];
    }
  }
}

// ---------------- host launcher ----------------
extern "C" void kernel_launch(void* const* d_in, const int* in_sizes, int n_in,
                              void* d_out, int out_size, void* d_ws, size_t ws_size,
                              hipStream_t stream) {
  (void)in_sizes; (void)n_in; (void)out_size; (void)ws_size;
  const float* x    = (const float*)d_in[0];
  const int*   mask = (const int*)  d_in[1];
  const float* wih  = (const float*)d_in[2];
  const float* whh  = (const float*)d_in[3];
  const float* bih  = (const float*)d_in[4];
  const float* bhh  = (const float*)d_in[5];
  const float* h0   = (const float*)d_in[6];
  const float* c0   = (const float*)d_in[7];
  float* out = (float*)d_out;

  char* ws = (char*)d_ws;
  size_t off = 0;
  auto alloc = [&](size_t bytes) -> void* {
    void* p = ws + off;
    off += (bytes + 255) & ~(size_t)255;
    return p;
  };
  unsigned short* xbf   = (unsigned short*)alloc((size_t)T_STEPS * BATCH * IDIM * 2);
  unsigned short* wihbf = (unsigned short*)alloc((size_t)NL * NGATES * IDIM * 2);
  unsigned short* whhbf = (unsigned short*)alloc((size_t)NL * NGATES * HDIM * 2);
  unsigned short* h0bf  = (unsigned short*)alloc((size_t)NL * HDIM * 2);
  float*          bsum  = (float*)alloc((size_t)NL * NGATES * 4);
  unsigned short* hbf   = (unsigned short*)alloc((size_t)NL * BATCH * HDIM * 2);
  float*          hf    = (float*)alloc((size_t)NL * BATCH * HDIM * 4);
  float*          cst   = (float*)alloc((size_t)NL * BATCH * HDIM * 4);
  float*          gates = (float*)alloc((size_t)BATCH * NGATES * 4);
  unsigned*       bars  = (unsigned*)alloc(256);

  cvt_bf16_kernel<<<4096, 256, 0, stream>>>(x,   xbf,   (size_t)T_STEPS * BATCH * IDIM);
  cvt_bf16_kernel<<<1024, 256, 0, stream>>>(wih, wihbf, (size_t)NL * NGATES * IDIM);
  cvt_bf16_kernel<<<1024, 256, 0, stream>>>(whh, whhbf, (size_t)NL * NGATES * HDIM);
  init_kernel<<<64, 256, 0, stream>>>(bih, bhh, h0, c0, bsum, h0bf, hbf, hf, cst, bars);

  lstm_persistent<<<NWG, NTHR, 0, stream>>>(
      xbf, mask, wihbf, whhbf, bsum, h0bf, c0,
      hbf, hf, cst, gates, bars + 0, bars + 1, out);
}
